// MulitHeadAttention_37357625540731
// MI455X (gfx1250) — matrix operand order repaired and checked by reading, not compiled, not run on hardware
//
#include <hip/hip_runtime.h>
#include <hip/hip_bf16.h>

// Fused MHA block for MI455X (gfx1250): bf16 WMMA GEMMs + TDM-staged flash
// attention + LN.  A/B operands use contiguous k = 16*half + s order.

#define HIDDEN 768
#define NHEADS 12
#define DHEAD  64
#define BATCH  8
#define SEQ    1024
#define MROWS  (BATCH * SEQ)

typedef __attribute__((ext_vector_type(16))) __bf16 v16bf;
typedef __attribute__((ext_vector_type(8)))  float  v8f;
typedef __attribute__((ext_vector_type(4)))  unsigned int v4u;
typedef __attribute__((ext_vector_type(8)))  int    v8i;
typedef __attribute__((ext_vector_type(4)))  int    v4i;

union FragBF { unsigned int u[8]; uint4 q2[2]; v16bf v; };
union FragF  { float f[8];        v8f  v; };

// f32 -> bf16 conversion: native packed cvt if the builtin exists, else RNE.
#if __has_builtin(__builtin_amdgcn_cvt_pk_bf16_f32)
__device__ __forceinline__ unsigned pack_bf2(float lo, float hi) {
  auto r = __builtin_amdgcn_cvt_pk_bf16_f32(lo, hi);
  return __builtin_bit_cast(unsigned, r);
}
__device__ __forceinline__ unsigned short f2bf(float f) {
  return (unsigned short)pack_bf2(f, 0.f);
}
#else
__device__ __forceinline__ unsigned short f2bf(float f) {
  union { float f; unsigned int u; } c; c.f = f;
  unsigned int u = c.u;
  return (unsigned short)((u + 0x7FFFu + ((u >> 16) & 1u)) >> 16);
}
__device__ __forceinline__ unsigned pack_bf2(float lo, float hi) {
  return (unsigned)f2bf(lo) | ((unsigned)f2bf(hi) << 16);
}
#endif

// low 32 bits of a generic pointer to an LDS object == LDS byte offset
__device__ __forceinline__ unsigned lds_lo(const void* p) {
  return (unsigned)(unsigned long long)(size_t)p;
}

// ---------------------------------------------------------------------------
// TDM: 2-D bf16 tile (64 x 64 elements) global -> LDS, with LDS row padding of
// 4 dwords every 32 dwords (=> 72-ushort LDS row stride).  6-arg builtin form.
// ---------------------------------------------------------------------------
__device__ __forceinline__ void tdm_load_tile64(unsigned lds_off, const void* gaddr,
                                                unsigned stride0 /*elements*/) {
  unsigned long long ga = (unsigned long long)(size_t)gaddr;
  v4u g0;
  g0.x = 1u;                                   // count=1, user descriptor
  g0.y = lds_off;                              // lds_addr
  g0.z = (unsigned)ga;                         // global_addr[31:0]
  g0.w = (unsigned)((ga >> 32) & 0x01FFFFFFull) | (2u << 30);  // addr[56:32]|type=2
  v8i g1;
  g1[0] = (int)((1u << 16) | (1u << 20) | (4u << 22) | (3u << 25));
          // data_size=2B, pad_enable, pad_interval=32dw, pad_amount=4dw
  g1[1] = (int)(64u << 16);                    // tensor_dim0 = 64
  g1[2] = (int)(64u << 16);                    // tensor_dim1 = 64
  g1[3] = (int)(64u << 16);                    // tile_dim0 = 64
  g1[4] = (int)64;                             // tile_dim1 = 64, tile_dim2 = 0
  g1[5] = (int)stride0;                        // tensor_dim0_stride
  g1[6] = 0;
  g1[7] = 0;
  v4i z4 = {0, 0, 0, 0};
  v8i z8 = {0, 0, 0, 0, 0, 0, 0, 0};
  __builtin_amdgcn_tensor_load_to_lds(g0, g1, z4, z4, z8, 0);
}

// ---------------------------------------------------------------------------
// Kernel 1: one QKV projection.  out = (x@w + bias)*oscale, bf16.
// VMODE=0 -> [B,H,S,D]; VMODE=1 -> [B,H,D,S] (transposed, for V so the
// attention kernel's [d][key] LDS tile is a straight TDM copy).
// Grid: (HIDDEN/64, MROWS/64), block 256 (8 waves), 64x64 tile per WG.
// ---------------------------------------------------------------------------
template <int VMODE>
__global__ __launch_bounds__(256)
void qkv_proj_kernel(const float* __restrict__ x, const float* __restrict__ w,
                     const float* __restrict__ bias, unsigned short* __restrict__ obf,
                     float oscale)
{
  __shared__ unsigned short xt[64][40];   // [m][k] bf16, 80B rows (16B-multiple)
  __shared__ unsigned short wtT[64][40];  // [n][k] bf16 (transposed weight tile)

  const int tid  = threadIdx.x;
  const int wid  = tid >> 5;
  const int lane = tid & 31;
  const int half = lane >> 4;
  const int lm   = lane & 15;

  const int m0 = blockIdx.y * 64;
  const int n0 = blockIdx.x * 64;

  const int mi  = wid & 3;           // 16-row subtile
  const int njb = (wid >> 2) * 32;   // 32-col strip

  FragF c0, c1;
  #pragma unroll
  for (int r = 0; r < 8; ++r) { c0.f[r] = 0.f; c1.f[r] = 0.f; }

  for (int kc = 0; kc < HIDDEN; kc += 32) {
    // stage x tile 64x32 as packed bf16 pairs (one ds_store_b32 per pair)
    for (int i = tid; i < 64 * 16; i += 256) {
      int r = i >> 4, c2 = i & 15;
      float2 xv = ((const float2*)(x + (size_t)(m0 + r) * HIDDEN + kc))[c2];
      *(unsigned*)&xt[r][c2 * 2] = pack_bf2(xv.x, xv.y);
    }
    // stage w tile 32x64 transposed -> wtT[n][k]
    for (int i = tid; i < 32 * 32; i += 256) {
      int r = i >> 5, c2 = i & 31;
      float2 wv = ((const float2*)(w + (size_t)(kc + r) * HIDDEN + n0))[c2];
      wtT[c2 * 2][r]     = f2bf(wv.x);
      wtT[c2 * 2 + 1][r] = f2bf(wv.y);
    }
    __syncthreads();

    FragBF a, b0, b1;
    const int arow = mi * 16 + lm;
    // A layout: u[j] = {K=2j+16h} -> 8 contiguous dwords in xt row m
    a.q2[0] = *(const uint4*)&xt[arow][16 * half];
    a.q2[1] = *(const uint4*)&xt[arow][16 * half + 8];
    // B layout: u[j] = {K=2j+16h} -> 8 contiguous dwords in wtT row n
    const int n1 = njb + lm, n2 = njb + 16 + lm;
    b0.q2[0] = *(const uint4*)&wtT[n1][16 * half];
    b0.q2[1] = *(const uint4*)&wtT[n1][16 * half + 8];
    b1.q2[0] = *(const uint4*)&wtT[n2][16 * half];
    b1.q2[1] = *(const uint4*)&wtT[n2][16 * half + 8];

    c0.v = __builtin_amdgcn_wmma_f32_16x16x32_bf16(false, a.v, false, b0.v, (short)0, c0.v, false, false);
    c1.v = __builtin_amdgcn_wmma_f32_16x16x32_bf16(false, a.v, false, b1.v, (short)0, c1.v, false, false);
    __syncthreads();
  }

  #pragma unroll
  for (int t = 0; t < 2; ++t) {
    FragF& cc = t ? c1 : c0;
    int nsub = njb + t * 16;
    #pragma unroll
    for (int r = 0; r < 8; ++r) {
      int m = m0 + mi * 16 + r + 8 * half;   // C layout: m = r + 8*half
      int n = n0 + nsub + lm;                // n = lane%16
      float vv = (cc.f[r] + bias[n]) * oscale;
      int b = m >> 10, s = m & 1023, h = n >> 6, d = n & 63;
      if (VMODE)
        obf[((size_t)((b * NHEADS + h) * DHEAD + d)) * SEQ + s] = f2bf(vv);
      else
        obf[((size_t)((b * NHEADS + h) * SEQ + s)) * DHEAD + d] = f2bf(vv);
    }
  }
}

// ---------------------------------------------------------------------------
// Kernel 2: flash attention per (b,h).  Grid: (SEQ/64, BATCH*NHEADS), block 128.
// K [key][d] and V-transposed [d][key] tiles staged by the Tensor Data Mover
// into double-buffered padded LDS tiles; wave 0 drives the TDM, TENSORcnt
// double-buffer pipeline.  Q pre-scaled by 1/sqrt(DHEAD) in kernel 1.
// ---------------------------------------------------------------------------
__global__ __launch_bounds__(128)
void attn_kernel(const unsigned short* __restrict__ Q,
                 const unsigned short* __restrict__ K,
                 const unsigned short* __restrict__ V,
                 float* __restrict__ ctx)
{
  __shared__ unsigned short kt[2][64][72];    // [buf][key][d], 144B rows
  __shared__ unsigned short vtT[2][64][72];   // [buf][d][key]
  __shared__ unsigned short pt[4][16][72];    // per-wave P tile [m][key]

  const int tid  = threadIdx.x;
  const int wid  = tid >> 5;
  const int lane = tid & 31;
  const int half = lane >> 4;
  const int lm   = lane & 15;

  const int bh = blockIdx.y;
  const int b = bh / NHEADS, h = bh % NHEADS;
  const unsigned short* q = Q + (size_t)bh * SEQ * DHEAD;
  const unsigned short* k = K + (size_t)bh * SEQ * DHEAD;
  const unsigned short* v = V + (size_t)bh * SEQ * DHEAD;  // [d][s] layout

  const int q0 = blockIdx.x * 64 + wid * 16;

  // Q fragments: K-dim = DHEAD(64), two 32-chunks, A layout, b128 global loads
  FragBF aq[2];
  {
    const uint4* q128 = (const uint4*)(q + (size_t)(q0 + lm) * DHEAD);
    aq[0].q2[0] = q128[2 * half];       // c=0, j<4 : d = 16h .. 16h+7
    aq[0].q2[1] = q128[2 * half + 1];   // c=0, j>=4: d = 16h+8 .. 16h+15
    aq[1].q2[0] = q128[4 + 2 * half];   // c=1, j<4
    aq[1].q2[1] = q128[5 + 2 * half];   // c=1, j>=4
  }

  float mrow[8], lrow[8];
  FragF acc[4];
  #pragma unroll
  for (int r = 0; r < 8; ++r) { mrow[r] = -3.0e38f; lrow[r] = 0.f; }
  #pragma unroll
  for (int t = 0; t < 4; ++t)
    #pragma unroll
    for (int r = 0; r < 8; ++r) acc[t].f[r] = 0.f;

  const unsigned ldsk[2] = { lds_lo(&kt[0][0][0]),  lds_lo(&kt[1][0][0]) };
  const unsigned ldsv[2] = { lds_lo(&vtT[0][0][0]), lds_lo(&vtT[1][0][0]) };
  const int NB = SEQ / 64;

  if (wid == 0) {                              // prime buffer 0
    tdm_load_tile64(ldsk[0], k, DHEAD);        // K rows: stride DHEAD
    tdm_load_tile64(ldsv[0], v, SEQ);          // Vt rows: stride SEQ
  }

  for (int i = 0; i < NB; ++i) {
    const int cur = i & 1;
    if (wid == 0) {
      if (i + 1 < NB) {
        tdm_load_tile64(ldsk[cur ^ 1], k + (size_t)(i + 1) * 64 * DHEAD, DHEAD);
        tdm_load_tile64(ldsv[cur ^ 1], v + (i + 1) * 64, SEQ);
        __builtin_amdgcn_s_wait_tensorcnt(2);  // current pair complete (in-order)
      } else {
        __builtin_amdgcn_s_wait_tensorcnt(0);
      }
    }
    __syncthreads();

    // scores: 4 key-subtiles of 16, K-dim = DHEAD via 2 WMMAs each
    FragF sc[4];
    #pragma unroll
    for (int t = 0; t < 4; ++t) {
      #pragma unroll
      for (int r = 0; r < 8; ++r) sc[t].f[r] = 0.f;
      const int key = t * 16 + lm;             // B layout: n = lane%16
      #pragma unroll
      for (int c = 0; c < 2; ++c) {
        FragBF bk;                             // B[k=d][n=key] = kt[key][d]
        bk.q2[0] = *(const uint4*)&kt[cur][key][c * 32 + 16 * half];
        bk.q2[1] = *(const uint4*)&kt[cur][key][c * 32 + 16 * half + 8];
        sc[t].v = __builtin_amdgcn_wmma_f32_16x16x32_bf16(false, aq[c].v, false, bk.v,
                                                          (short)0, sc[t].v, false, false);
      }
    }

    // online softmax (16-lane xor-shuffle row reductions)
    #pragma unroll
    for (int r = 0; r < 8; ++r) {
      float rmax = fmaxf(fmaxf(sc[0].f[r], sc[1].f[r]), fmaxf(sc[2].f[r], sc[3].f[r]));
      #pragma unroll
      for (int msk = 1; msk < 16; msk <<= 1)
        rmax = fmaxf(rmax, __shfl_xor(rmax, msk, 32));
      float mnew = fmaxf(mrow[r], rmax);
      float al   = __expf(mrow[r] - mnew);
      float rsum = 0.f;
      #pragma unroll
      for (int t = 0; t < 4; ++t) {
        float p = __expf(sc[t].f[r] - mnew);
        rsum += p;
        pt[wid][r + 8 * half][t * 16 + lm] = f2bf(p);   // C layout -> LDS [m][key]
      }
      #pragma unroll
      for (int msk = 1; msk < 16; msk <<= 1)
        rsum += __shfl_xor(rsum, msk, 32);
      lrow[r] = lrow[r] * al + rsum;
      mrow[r] = mnew;
      #pragma unroll
      for (int t = 0; t < 4; ++t) acc[t].f[r] *= al;
    }

    // ctx += P @ V   (K-dim = 64 keys in two 32-chunks)
    #pragma unroll
    for (int c = 0; c < 2; ++c) {
      FragBF ap;                               // A layout from per-wave P tile
      ap.q2[0] = *(const uint4*)&pt[wid][lm][c * 32 + 16 * half];
      ap.q2[1] = *(const uint4*)&pt[wid][lm][c * 32 + 16 * half + 8];
      #pragma unroll
      for (int t = 0; t < 4; ++t) {
        FragBF bv;                             // B[k=key][n=d] = vtT[d][key]
        bv.q2[0] = *(const uint4*)&vtT[cur][t * 16 + lm][c * 32 + 16 * half];
        bv.q2[1] = *(const uint4*)&vtT[cur][t * 16 + lm][c * 32 + 16 * half + 8];
        acc[t].v = __builtin_amdgcn_wmma_f32_16x16x32_bf16(false, ap.v, false, bv.v,
                                                           (short)0, acc[t].v, false, false);
      }
    }
    __syncthreads();
  }

  // normalize (one reciprocal per row) and write ctx as [B,S,HIDDEN] f32
  #pragma unroll
  for (int r = 0; r < 8; ++r) {
    const float inv = 1.0f / lrow[r];
    const int s = q0 + r + 8 * half;
    float* orow = ctx + ((size_t)(b * SEQ + s)) * HIDDEN + h * DHEAD;
    #pragma unroll
    for (int t = 0; t < 4; ++t)
      orow[t * 16 + lm] = acc[t].f[r] * inv;
  }
}

// ---------------------------------------------------------------------------
// Kernel 3: out = LayerNorm(x + 2*ctx) * gamma + beta.  One WG (192 thr) per
// row; one float4 per thread (768 = 192*4).
// ---------------------------------------------------------------------------
__global__ __launch_bounds__(192)
void ln_kernel(const float* __restrict__ x, const float* __restrict__ ctx,
               const float* __restrict__ gamma, const float* __restrict__ beta,
               float* __restrict__ out)
{
  __shared__ float s_sum[6], s_sq[6];
  const int row = blockIdx.x;
  const size_t base = (size_t)row * HIDDEN;
  const int tid = threadIdx.x;

  const float4 xr = ((const float4*)(x + base))[tid];
  const float4 cr = ((const float4*)(ctx + base))[tid];
  float4 y;
  y.x = xr.x + 2.0f * cr.x;
  y.y = xr.y + 2.0f * cr.y;
  y.z = xr.z + 2.0f * cr.z;
  y.w = xr.w + 2.0f * cr.w;

  float sum = y.x + y.y + y.z + y.w;
  float sq  = y.x * y.x + y.y * y.y + y.z * y.z + y.w * y.w;
  #pragma unroll
  for (int msk = 1; msk < 32; msk <<= 1) {
    sum += __shfl_xor(sum, msk, 32);
    sq  += __shfl_xor(sq,  msk, 32);
  }
  if ((tid & 31) == 0) { s_sum[tid >> 5] = sum; s_sq[tid >> 5] = sq; }
  __syncthreads();
  float ts = 0.f, tq = 0.f;
  #pragma unroll
  for (int wv = 0; wv < 6; ++wv) { ts += s_sum[wv]; tq += s_sq[wv]; }
  const float mu  = ts * (1.0f / HIDDEN);
  const float var = tq * (1.0f / HIDDEN) - mu * mu;
  const float inv = rsqrtf(var + 1e-6f);

  const float4 g  = ((const float4*)gamma)[tid];
  const float4 bb = ((const float4*)beta)[tid];
  float4 o;
  o.x = (y.x - mu) * inv * g.x + bb.x;
  o.y = (y.y - mu) * inv * g.y + bb.y;
  o.z = (y.z - mu) * inv * g.z + bb.z;
  o.w = (y.w - mu) * inv * g.w + bb.w;
  ((float4*)(out + base))[tid] = o;
}

// ---------------------------------------------------------------------------
extern "C" void kernel_launch(void* const* d_in, const int* in_sizes, int n_in,
                              void* d_out, int out_size, void* d_ws, size_t ws_size,
                              hipStream_t stream) {
  const float* x     = (const float*)d_in[0];
  const float* wq    = (const float*)d_in[1];
  const float* bq    = (const float*)d_in[2];
  const float* wk    = (const float*)d_in[3];
  const float* bk    = (const float*)d_in[4];
  const float* wv    = (const float*)d_in[5];
  const float* bv    = (const float*)d_in[6];
  const float* gamma = (const float*)d_in[7];
  const float* beta  = (const float*)d_in[8];

  const size_t QKV_ELEMS = (size_t)BATCH * NHEADS * SEQ * DHEAD;  // 6291456
  unsigned short* qws = (unsigned short*)d_ws;
  unsigned short* kws = qws + QKV_ELEMS;
  unsigned short* vws = kws + QKV_ELEMS;
  float* ctx = (float*)(vws + QKV_ELEMS);

  dim3 gProj(HIDDEN / 64, MROWS / 64);     // (12, 128)
  const float qscale = 0.125f;             // 1/sqrt(DHEAD), folded into Q
  qkv_proj_kernel<0><<<gProj, 256, 0, stream>>>(x, wq, bq, qws, qscale);
  qkv_proj_kernel<0><<<gProj, 256, 0, stream>>>(x, wk, bk, kws, 1.0f);
  qkv_proj_kernel<1><<<gProj, 256, 0, stream>>>(x, wv, bv, vws, 1.0f);  // V transposed

  attn_kernel<<<dim3(SEQ / 64, BATCH * NHEADS), 128, 0, stream>>>(qws, kws, vws, ctx);

  ln_kernel<<<MROWS, 192, 0, stream>>>(x, ctx, gamma, beta, (float*)d_out);
}
